// LDEPool1d_9028021256468
// MI455X (gfx1250) — matrix operand order repaired and checked by reading, not compiled, not run on hardware
//
#include <hip/hip_runtime.h>

typedef __attribute__((ext_vector_type(16))) _Float16 v16h;
typedef __attribute__((ext_vector_type(8)))  _Float16 v8h;
typedef __attribute__((ext_vector_type(8)))  float    v8f;

#define B_  8
#define D_  256
#define T_  1024
#define K_  64
#define LSTR 264   // pass-A LDS row stride (halfs): 528B = 132 dwords -> bank-conflict-free, 16B aligned
#define PSTR 40    // pass-B LDS row stride (halfs): 80B = 20 dwords, 16B aligned

// ---------------- prep kernels ----------------

__global__ void prep_mu_kernel(const float* __restrict__ mu,
                               const float* __restrict__ prec,
                               float* __restrict__ munorm,
                               float* __restrict__ prec2) {
  int k = threadIdx.x;          // 64 threads
  float s = 0.f;
  for (int d = 0; d < D_; ++d) { float m = mu[k * D_ + d]; s += m * m; }
  munorm[k] = s;
  prec2[k]  = prec[k] * prec[k];
}

__global__ void prep_xnorm_kernel(const float* __restrict__ x,
                                  float* __restrict__ xnorm) {
  int id = blockIdx.x * blockDim.x + threadIdx.x;   // b*T + t
  int b = id >> 10, t = id & (T_ - 1);
  const float* p = x + (size_t)b * D_ * T_ + t;
  float s = 0.f;
  for (int d = 0; d < D_; ++d) { float v = p[(size_t)d * T_]; s += v * v; }
  xnorm[id] = s;
}

// ---------------- pass A: scores + softmax -> r (f16) ----------------
// grid: B*8 blocks, 256 threads (8 waves); wave w owns t-rows [t0+16w, t0+16w+16)

__global__ void passA_kernel(const float* __restrict__ x,
                             const float* __restrict__ mu,
                             const float* __restrict__ xnorm,
                             const float* __restrict__ munorm,
                             const float* __restrict__ prec2,
                             _Float16* __restrict__ r16) {
  extern __shared__ _Float16 smem[];
  _Float16* mus = smem;               // [K_][LSTR]   f16 copy of mu
  _Float16* xs  = smem + K_ * LSTR;   // [128][LSTR]  f16 x-slice, [t][d]

  const int tid = threadIdx.x;
  const int b   = blockIdx.x >> 3;
  const int t0  = (blockIdx.x & 7) * 128;

  // stage mu: 64x256 f32 -> f16 (coalesced, d fastest)
  #pragma unroll
  for (int i = 0; i < (K_ * D_) / 256; ++i) {
    int e = tid + 256 * i;
    int k = e >> 8, d = e & 255;
    mus[k * LSTR + d] = (_Float16)mu[e];
  }
  // stage x slice: 128 t x 256 d, transposed into xs[t][d] (t contiguous in memory)
  #pragma unroll
  for (int i = 0; i < (128 * D_) / 256; ++i) {
    int e = tid + 256 * i;
    int d = e >> 7, tt = e & 127;
    xs[tt * LSTR + d] = (_Float16)x[((size_t)b * D_ + d) * T_ + t0 + tt];
  }
  __syncthreads();

  const int w   = tid >> 5;
  const int l   = tid & 31;
  const int col = l & 15;      // WMMA B/C column, A row
  const int hi  = l >> 4;      // lane half
  const int trow = w * 16;

  v8f c[4];
  #pragma unroll
  for (int kt = 0; kt < 4; ++kt) c[kt] = (v8f){0.f,0.f,0.f,0.f,0.f,0.f,0.f,0.f};

  // S = x_tile (16x256) * mu^T (256x64), f16 inputs / f32 accum
  #pragma unroll
  for (int dc = 0; dc < 8; ++dc) {
    const _Float16* ap = xs + (trow + col) * LSTR + dc * 32 + 8 * hi;
    v8h alo = *(const v8h*)ap;            // kk = 8*hi + 0..7
    v8h ahi = *(const v8h*)(ap + 16);     // kk = 16 + 8*hi + 0..7
    v16h a;
    #pragma unroll
    for (int i = 0; i < 8; ++i) { a[i] = alo[i]; a[i + 8] = ahi[i]; }
    #pragma unroll
    for (int kt = 0; kt < 4; ++kt) {
      const _Float16* bp = mus + (kt * 16 + col) * LSTR + dc * 32 + 8 * hi;
      v8h blo = *(const v8h*)bp;
      v8h bhi = *(const v8h*)(bp + 16);
      v16h bm;
      #pragma unroll
      for (int i = 0; i < 8; ++i) { bm[i] = blo[i]; bm[i + 8] = bhi[i]; }
      c[kt] = __builtin_amdgcn_wmma_f32_16x16x32_f16(false, a, false, bm,
                                                     (short)0, c[kt], false, false);
    }
  }

  // llk = -prec^2 * (|x|^2 - 2S + |mu|^2); softmax over k (64) per t-row
  float xn[8];
  #pragma unroll
  for (int v = 0; v < 8; ++v)
    xn[v] = xnorm[b * T_ + t0 + trow + 8 * hi + v];

  float ll[4][8];
  #pragma unroll
  for (int kt = 0; kt < 4; ++kt) {
    int k = kt * 16 + col;
    float mn = munorm[k], pp = prec2[k];
    #pragma unroll
    for (int v = 0; v < 8; ++v)
      ll[kt][v] = -pp * (xn[v] - 2.f * c[kt][v] + mn);
  }

  #pragma unroll
  for (int v = 0; v < 8; ++v) {
    float rm = fmaxf(fmaxf(ll[0][v], ll[1][v]), fmaxf(ll[2][v], ll[3][v]));
    rm = fmaxf(rm, __shfl_xor(rm, 1, 32));
    rm = fmaxf(rm, __shfl_xor(rm, 2, 32));
    rm = fmaxf(rm, __shfl_xor(rm, 4, 32));
    rm = fmaxf(rm, __shfl_xor(rm, 8, 32));   // xor<16 stays in the half: 16-lane row reduce
    float e0 = __expf(ll[0][v] - rm), e1 = __expf(ll[1][v] - rm);
    float e2 = __expf(ll[2][v] - rm), e3 = __expf(ll[3][v] - rm);
    float rs = e0 + e1 + e2 + e3;
    rs += __shfl_xor(rs, 1, 32);
    rs += __shfl_xor(rs, 2, 32);
    rs += __shfl_xor(rs, 4, 32);
    rs += __shfl_xor(rs, 8, 32);
    float inv = 1.f / rs;
    int t = t0 + trow + 8 * hi + v;
    size_t base = ((size_t)b * T_ + t) * K_;
    r16[base +  0 + col] = (_Float16)(e0 * inv);
    r16[base + 16 + col] = (_Float16)(e1 * inv);
    r16[base + 32 + col] = (_Float16)(e2 * inv);
    r16[base + 48 + col] = (_Float16)(e3 * inv);
  }
}

// ---------------- pass B: F = r^T x, N = sum_t r, pool = (F - N*mu)/(N+eps) ----------------
// grid: B * (K/16) * (D/16) = 512 blocks, 1 wave each

__global__ void passB_kernel(const float* __restrict__ x,
                             const float* __restrict__ mu,
                             const _Float16* __restrict__ r16,
                             float* __restrict__ out) {
  __shared__ _Float16 as[16 * PSTR];   // A: r tile, [k][t]
  __shared__ _Float16 bs[16 * PSTR];   // B: x tile, [d][t]
  __shared__ float nsh[16];

  const int b  = blockIdx.x >> 6;
  const int kt = (blockIdx.x >> 4) & 3;
  const int dt = blockIdx.x & 15;
  const int k0 = kt * 16, d0 = dt * 16;
  const int l = threadIdx.x;
  const int col = l & 15, hi = l >> 4;

  v8f c = (v8f){0.f,0.f,0.f,0.f,0.f,0.f,0.f,0.f};
  float nacc = 0.f;

  for (int tc = 0; tc < T_ / 32; ++tc) {
    // stage A: r16[b, t, k0..k0+15], t in [32tc, 32tc+32) -> as[k][t]
    #pragma unroll
    for (int i = 0; i < 16; ++i) {
      int e = l + 32 * i;
      int k = e & 15, t = e >> 4;
      as[k * PSTR + t] = r16[((size_t)b * T_ + tc * 32 + t) * K_ + k0 + k];
    }
    // stage B: x[b, d0..d0+15, t] -> bs[d][t] f16 (coalesced: t fastest)
    #pragma unroll
    for (int i = 0; i < 16; ++i) {
      int e = l + 32 * i;
      int t = e & 31, d = e >> 5;
      bs[d * PSTR + t] = (_Float16)x[((size_t)b * D_ + d0 + d) * T_ + tc * 32 + t];
    }
    __syncthreads();

    const _Float16* ap = as + col * PSTR + 8 * hi;
    v8h alo = *(const v8h*)ap;
    v8h ahi = *(const v8h*)(ap + 16);
    v16h a;
    #pragma unroll
    for (int i = 0; i < 8; ++i) { a[i] = alo[i]; a[i + 8] = ahi[i]; }
    // fold N = sum_t r while the r tile is in registers (lane pair l/l+16 covers all 32 t)
    #pragma unroll
    for (int i = 0; i < 8; ++i) nacc += (float)alo[i] + (float)ahi[i];

    const _Float16* bp = bs + col * PSTR + 8 * hi;
    v8h blo = *(const v8h*)bp;
    v8h bhi = *(const v8h*)(bp + 16);
    v16h bm;
    #pragma unroll
    for (int i = 0; i < 8; ++i) { bm[i] = blo[i]; bm[i + 8] = bhi[i]; }
    c = __builtin_amdgcn_wmma_f32_16x16x32_f16(false, a, false, bm,
                                               (short)0, c, false, false);
    __syncthreads();
  }

  nacc += __shfl_xor(nacc, 16, 32);
  if (l < 16) nsh[l] = nacc;      // N for k = k0 + l
  __syncthreads();

  #pragma unroll
  for (int v = 0; v < 8; ++v) {
    int m = v + 8 * hi;                   // C row
    int k = k0 + m, d = d0 + col;
    float Nsum = nsh[m];
    float muv  = mu[k * D_ + d];
    out[(size_t)b * (K_ * D_) + (size_t)k * D_ + d] =
        (c[v] - Nsum * muv) / (Nsum + 1e-9f);
  }
}

// ---------------- launcher ----------------

extern "C" void kernel_launch(void* const* d_in, const int* in_sizes, int n_in,
                              void* d_out, int out_size, void* d_ws, size_t ws_size,
                              hipStream_t stream) {
  const float* x    = (const float*)d_in[0];   // (B, D, T)
  const float* mu   = (const float*)d_in[1];   // (K, D)
  const float* prec = (const float*)d_in[2];   // (K,)
  float* out = (float*)d_out;                  // (B, K*D)

  char* ws = (char*)d_ws;
  float*    xnorm  = (float*)(ws);             // B*T floats   = 32768 B
  float*    munorm = (float*)(ws + 32768);     // 64 floats
  float*    prec2  = (float*)(ws + 33024);     // 64 floats
  _Float16* r16    = (_Float16*)(ws + 33280);  // B*T*K halfs  = 1 MB

  prep_mu_kernel<<<1, 64, 0, stream>>>(mu, prec, munorm, prec2);
  prep_xnorm_kernel<<<(B_ * T_) / 256, 256, 0, stream>>>(x, xnorm);

  size_t smemA = (size_t)(K_ + 128) * LSTR * sizeof(_Float16);  // ~99 KB dynamic LDS (WGP has 320 KB)
  passA_kernel<<<B_ * 8, 256, smemA, stream>>>(x, mu, xnorm, munorm, prec2, r16);

  passB_kernel<<<B_ * 4 * 16, 32, 0, stream>>>(x, mu, r16, out);
}
